// MultiHeadAttention_8340826489626
// MI455X (gfx1250) — matrix operand order repaired and checked by reading, not compiled, not run on hardware
//
#include <hip/hip_runtime.h>
#include <cstdint>

typedef __bf16 bf16_t;
typedef __attribute__((ext_vector_type(16))) __bf16 v16bf;
typedef __attribute__((ext_vector_type(8)))  __bf16 v8bf;
typedef __attribute__((ext_vector_type(8)))  float  v8f;

union V16U { v16bf v; v8bf h[2]; };

// A-matrix fragment (16xK tile, K contiguous, row stride `stride` elems).
// Lane l<16: row l, K = {0..7, 16..23}; lane l>=16: row l-16, K = {8..15, 24..31}.
__device__ __forceinline__ v16bf frag_a(const bf16_t* base, int stride, int lane) {
    const bf16_t* p = base + (size_t)(lane & 15) * stride + ((lane >> 4) << 3);
    V16U u;
    u.h[0] = *(const v8bf*)(p);
    u.h[1] = *(const v8bf*)(p + 16);
    return u.v;
}

// B-matrix fragment from B^T layout ([N][K], K contiguous, row stride `stride`).
__device__ __forceinline__ v16bf frag_b(const bf16_t* base, int stride, int lane) {
    const bf16_t* p = base + (size_t)(lane & 15) * stride + ((lane >> 4) << 3);
    V16U u;
    u.h[0] = *(const v8bf*)(p);
    u.h[1] = *(const v8bf*)(p + 16);
    return u.v;
}

#define WMMA_BF16(A, B, C) \
    __builtin_amdgcn_wmma_f32_16x16x32_bf16(false, (A), false, (B), (short)0, (C), false, false)

// 16-byte async global->LDS copy (per-lane addresses), tracked by ASYNCcnt.
__device__ __forceinline__ void async_copy16(void* ldsp, const void* gp) {
    unsigned lds = (unsigned)(uintptr_t)ldsp;          // low 32 bits = LDS offset
    unsigned long long g = (unsigned long long)(uintptr_t)gp;
    asm volatile("global_load_async_to_lds_b128 %0, %1, off"
                 : : "v"(lds), "v"(g) : "memory");
}
__device__ __forceinline__ void wait_async0() {
    asm volatile("s_wait_asynccnt 0x0" ::: "memory");
}

// ---------------------------------------------------------------------------
// Kernel 0a: x (f32 [4096,1024]) -> bf16, same layout.
// ---------------------------------------------------------------------------
__global__ __launch_bounds__(256) void k_cvt_x(const float* __restrict__ src,
                                               bf16_t* __restrict__ dst) {
    size_t i = ((size_t)blockIdx.x * 256 + threadIdx.x) * 8;
    float4 a = *(const float4*)(src + i);
    float4 b = *(const float4*)(src + i + 4);
    union { bf16_t h[8]; uint4 u; } p;
    p.h[0] = (bf16_t)a.x; p.h[1] = (bf16_t)a.y;
    p.h[2] = (bf16_t)a.z; p.h[3] = (bf16_t)a.w;
    p.h[4] = (bf16_t)b.x; p.h[5] = (bf16_t)b.y;
    p.h[6] = (bf16_t)b.z; p.h[7] = (bf16_t)b.w;
    *(uint4*)(dst + i) = p.u;
}

// ---------------------------------------------------------------------------
// Kernel 0b: W (f32 [1024 k][1024 n]) -> W^T bf16 [1024 n][1024 k].
// z selects {Wq, Wk, Wv, Wo}.
// ---------------------------------------------------------------------------
__global__ __launch_bounds__(256) void k_cvt_wT(
    const float* __restrict__ Wq, const float* __restrict__ Wk,
    const float* __restrict__ Wv, const float* __restrict__ Wo,
    bf16_t* __restrict__ t0, bf16_t* __restrict__ t1,
    bf16_t* __restrict__ t2, bf16_t* __restrict__ t3)
{
    const int z = blockIdx.z;
    const float* W = (z == 0) ? Wq : (z == 1) ? Wk : (z == 2) ? Wv : Wo;
    bf16_t* T = (z == 0) ? t0 : (z == 1) ? t1 : (z == 2) ? t2 : t3;
    int t = blockIdx.x * 256 + threadIdx.x;
    int n  = t & 1023;
    int kb = (t >> 10) * 8;
    union { bf16_t h[8]; uint4 u; } p;
    #pragma unroll
    for (int j = 0; j < 8; ++j)
        p.h[j] = (bf16_t)W[(size_t)(kb + j) * 1024 + n];
    *(uint4*)(T + (size_t)n * 1024 + kb) = p.u;
}

// ---------------------------------------------------------------------------
// GEMM: A[4096,1024] bf16 @ Bt^T (Bt is [1024 n][1024 k] bf16).
// 128x128 block tile, 8 waves, K-step 64, async double-buffered LDS staging.
// MODE 0: bf16 out head-major [B][H][S][64] (used for q and k)
// MODE 2: bf16 out transposed [B][H][64][S]  (v^T)
// MODE 3: f32 out [4096][1024] + bias
// ---------------------------------------------------------------------------
template <int MODE>
__global__ __launch_bounds__(256) void k_gemm(
    const bf16_t* __restrict__ A, const bf16_t* __restrict__ Bt,
    const float* __restrict__ bias, void* __restrict__ Cout)
{
    __shared__ __align__(16) bf16_t As[2][128][72];
    __shared__ __align__(16) bf16_t Bs[2][128][72];

    const int m0 = blockIdx.x * 128;
    const int n0 = blockIdx.y * 128;
    const int tid = threadIdx.x, lane = tid & 31, wave = tid >> 5;
    const int wm = (wave >> 2) * 64;   // 0 | 64
    const int wn = (wave & 3) * 32;    // 0..96

    auto stage = [&](int buf, int k0) {
        #pragma unroll
        for (int i = 0; i < 4; ++i) {
            int ch  = tid + i * 256;          // 0..1023 (16B chunks)
            int row = ch >> 3;                // 0..127
            int col = (ch & 7) * 8;           // 0..56
            async_copy16(&As[buf][row][col], A  + (size_t)(m0 + row) * 1024 + k0 + col);
            async_copy16(&Bs[buf][row][col], Bt + (size_t)(n0 + row) * 1024 + k0 + col);
        }
    };

    v8f acc[4][2] = {};
    stage(0, 0);
    for (int k0 = 0; k0 < 1024; k0 += 64) {
        const int cur = (k0 >> 6) & 1;
        wait_async0();
        __syncthreads();
        if (k0 + 64 < 1024) stage(cur ^ 1, k0 + 64);
        #pragma unroll
        for (int mt = 0; mt < 4; ++mt) {
            v16bf a0 = frag_a(&As[cur][wm + mt * 16][0],  72, lane);
            v16bf a1 = frag_a(&As[cur][wm + mt * 16][32], 72, lane);
            #pragma unroll
            for (int nt = 0; nt < 2; ++nt) {
                v16bf b0 = frag_b(&Bs[cur][wn + nt * 16][0],  72, lane);
                v16bf b1 = frag_b(&Bs[cur][wn + nt * 16][32], 72, lane);
                acc[mt][nt] = WMMA_BF16(a0, b0, acc[mt][nt]);
                acc[mt][nt] = WMMA_BF16(a1, b1, acc[mt][nt]);
            }
        }
    }

    const int lhi = (lane >> 4) * 8;
    const int ln  = lane & 15;
    if constexpr (MODE == 0) {
        bf16_t* dst = (bf16_t*)Cout;
        #pragma unroll
        for (int mt = 0; mt < 4; ++mt)
            #pragma unroll
            for (int nt = 0; nt < 2; ++nt)
                #pragma unroll
                for (int r = 0; r < 8; ++r) {
                    int m = m0 + wm + mt * 16 + r + lhi;   // b*S + s
                    int n = n0 + wn + nt * 16 + ln;        // h*64 + d
                    int b = m >> 11, s = m & 2047, h = n >> 6, d = n & 63;
                    dst[(((size_t)b * 16 + h) * 2048 + s) * 64 + d] = (bf16_t)acc[mt][nt][r];
                }
    } else if constexpr (MODE == 2) {
        bf16_t* dst = (bf16_t*)Cout;
        #pragma unroll
        for (int mt = 0; mt < 4; ++mt)
            #pragma unroll
            for (int nt = 0; nt < 2; ++nt)
                #pragma unroll
                for (int r = 0; r < 8; ++r) {
                    int m = m0 + wm + mt * 16 + r + lhi;
                    int n = n0 + wn + nt * 16 + ln;
                    int b = m >> 11, s = m & 2047, h = n >> 6, d = n & 63;
                    dst[(((size_t)b * 16 + h) * 64 + d) * 2048 + s] = (bf16_t)acc[mt][nt][r];
                }
    } else {
        float* dst = (float*)Cout;
        #pragma unroll
        for (int nt = 0; nt < 2; ++nt) {
            int n = n0 + wn + nt * 16 + ln;
            float bv = bias[n];
            #pragma unroll
            for (int mt = 0; mt < 4; ++mt)
                #pragma unroll
                for (int r = 0; r < 8; ++r) {
                    int m = m0 + wm + mt * 16 + r + lhi;
                    dst[(size_t)m * 1024 + n] = acc[mt][nt][r] + bv;
                }
        }
    }
}

// ---------------------------------------------------------------------------
// Flash attention.  Per (b,h): Q[2048,64], K[2048,64], V^T[64,2048] (bf16).
// Block = 128 q rows (8 waves x 16 rows); KV tiles of 64 keys, async
// double-buffered into LDS.
// Scores computed TRANSPOSED (S^T = K Q^T) so softmax stats are per-lane:
// lane = query, VGPR = key.  Only shfl_xor(16) combines + 8 alpha broadcasts.
// Output bf16 attn[b][s][h*64+d].
// ---------------------------------------------------------------------------
__global__ __launch_bounds__(256) void k_attn(
    const bf16_t* __restrict__ qin, const bf16_t* __restrict__ kin,
    const bf16_t* __restrict__ vT, bf16_t* __restrict__ attn)
{
    const int qt = blockIdx.x, h = blockIdx.y, b = blockIdx.z;
    const int tid = threadIdx.x, lane = tid & 31, wave = tid >> 5;
    const int lhi = (lane >> 4) * 8;
    const int ln  = lane & 15;

    const size_t bh = (size_t)b * 16 + h;
    const bf16_t* Q = qin + bh * 2048 * 64;
    const bf16_t* K = kin + bh * 2048 * 64;
    const bf16_t* V = vT  + bh * 64 * 2048;
    const int q0 = qt * 128 + wave * 16;

    __shared__ __align__(16) bf16_t Ks[2][64][72];   // [key][dim]
    __shared__ __align__(16) bf16_t Vs[2][64][72];   // [dim][key]
    __shared__ __align__(16) bf16_t Ps[8][16][72];   // per-wave P strip [query][key]

    // Q as B-fragments (lane = query, K-dim = head dim, contiguous in global)
    v16bf qb0 = frag_b(Q + (size_t)q0 * 64 + 0,  64, lane);
    v16bf qb1 = frag_b(Q + (size_t)q0 * 64 + 32, 64, lane);

    auto stageKV = [&](int buf, int kv0) {
        int r = tid >> 2;            // 0..63
        int c = (tid & 3) * 16;      // 0,16,32,48
        async_copy16(&Ks[buf][r][c],     K + (size_t)(kv0 + r) * 64 + c);
        async_copy16(&Ks[buf][r][c + 8], K + (size_t)(kv0 + r) * 64 + c + 8);
        async_copy16(&Vs[buf][r][c],     V + (size_t)r * 2048 + kv0 + c);
        async_copy16(&Vs[buf][r][c + 8], V + (size_t)r * 2048 + kv0 + c + 8);
    };

    v8f o[4] = {};
    float m_run = -1e30f, l_run = 0.0f;
    const float scale = 0.125f;  // 1/sqrt(64)

    stageKV(0, 0);
    for (int kv0 = 0; kv0 < 2048; kv0 += 64) {
        const int cur = (kv0 >> 6) & 1;
        wait_async0();
        __syncthreads();
        if (kv0 + 64 < 2048) stageKV(cur ^ 1, kv0 + 64);

        // S^T tiles [key][query]: A = K-tile (16 keys x 64 dims), B = Q
        v8f sacc[4];
        #pragma unroll
        for (int nt = 0; nt < 4; ++nt) {
            v8f s = {};
            s = WMMA_BF16(frag_a(&Ks[cur][nt * 16][0],  72, lane), qb0, s);
            s = WMMA_BF16(frag_a(&Ks[cur][nt * 16][32], 72, lane), qb1, s);
            sacc[nt] = s;
        }

        // per-lane (per-query) online softmax over this lane's 32 keys
        float mx = sacc[0][0];
        #pragma unroll
        for (int nt = 0; nt < 4; ++nt)
            #pragma unroll
            for (int r = 0; r < 8; ++r)
                mx = fmaxf(mx, sacc[nt][r]);
        mx = fmaxf(mx, __shfl_xor(mx, 16, 32));   // combine other half's keys
        float mnew  = fmaxf(m_run, mx * scale);
        float alpha = __expf(m_run - mnew);
        m_run = mnew;

        float sum = 0.0f;
        #pragma unroll
        for (int nt = 0; nt < 4; ++nt)
            #pragma unroll
            for (int r = 0; r < 8; ++r) {
                float p = __expf(sacc[nt][r] * scale - mnew);
                sum += p;
                // P^T element (key nt*16+r+lhi, query ln) -> Ps[query][key]
                Ps[wave][ln][nt * 16 + r + lhi] = (bf16_t)p;
            }
        sum += __shfl_xor(sum, 16, 32);
        l_run = l_run * alpha + sum;
        asm volatile("s_wait_dscnt 0x0" ::: "memory");

        // broadcast alpha of query (r+lhi) to all lanes, rescale O rows
        float arow[8];
        #pragma unroll
        for (int r = 0; r < 8; ++r)
            arow[r] = __shfl(alpha, r + lhi, 32);
        #pragma unroll
        for (int nt = 0; nt < 4; ++nt)
            #pragma unroll
            for (int r = 0; r < 8; ++r)
                o[nt][r] *= arow[r];

        // O += P V : A = P [query][key] from strip, B = V^T [dim][key]
        v16bf pa0 = frag_a(&Ps[wave][0][0],  72, lane);
        v16bf pa1 = frag_a(&Ps[wave][0][32], 72, lane);
        #pragma unroll
        for (int nt = 0; nt < 4; ++nt) {
            o[nt] = WMMA_BF16(pa0, frag_b(&Vs[cur][nt * 16][0],  72, lane), o[nt]);
            o[nt] = WMMA_BF16(pa1, frag_b(&Vs[cur][nt * 16][32], 72, lane), o[nt]);
        }
    }

    float lr[8];
    #pragma unroll
    for (int r = 0; r < 8; ++r)
        lr[r] = __shfl(l_run, r + lhi, 32);
    #pragma unroll
    for (int nt = 0; nt < 4; ++nt)
        #pragma unroll
        for (int r = 0; r < 8; ++r) {
            int s = q0 + r + lhi;
            int d = nt * 16 + ln;
            attn[((size_t)b * 2048 + s) * 1024 + h * 64 + d] =
                (bf16_t)(o[nt][r] / lr[r]);
        }
}

// ---------------------------------------------------------------------------
extern "C" void kernel_launch(void* const* d_in, const int* in_sizes, int n_in,
                              void* d_out, int out_size, void* d_ws, size_t ws_size,
                              hipStream_t stream) {
    const float* x  = (const float*)d_in[0];
    const float* Wq = (const float*)d_in[1];
    const float* Wk = (const float*)d_in[2];
    const float* Wv = (const float*)d_in[3];
    const float* Wo = (const float*)d_in[4];
    const float* bo = (const float*)d_in[5];
    float* out = (float*)d_out;

    const size_t MN  = (size_t)4096 * 1024;   // 4M
    const size_t WSZ = (size_t)1024 * 1024;   // 1M
    bf16_t* xb  = (bf16_t*)d_ws;              // x bf16
    bf16_t* wqt = xb  + MN;                   // W^T bf16
    bf16_t* wkt = wqt + WSZ;
    bf16_t* wvt = wkt + WSZ;
    bf16_t* wot = wvt + WSZ;
    bf16_t* qb  = wot + WSZ;                  // q  [B][H][S][64]
    bf16_t* kb  = qb  + MN;                   // k  [B][H][S][64]
    bf16_t* vTb = kb  + MN;                   // v^T[B][H][64][S]
    bf16_t* ab  = vTb + MN;                   // attn [4096][1024]

    k_cvt_x<<<dim3(2048), dim3(256), 0, stream>>>(x, xb);
    k_cvt_wT<<<dim3(512, 1, 4), dim3(256), 0, stream>>>(Wq, Wk, Wv, Wo,
                                                        wqt, wkt, wvt, wot);

    dim3 gg(32, 8, 1);
    k_gemm<0><<<gg, dim3(256), 0, stream>>>(xb, wqt, nullptr, qb);
    k_gemm<0><<<gg, dim3(256), 0, stream>>>(xb, wkt, nullptr, kb);
    k_gemm<2><<<gg, dim3(256), 0, stream>>>(xb, wvt, nullptr, vTb);

    k_attn<<<dim3(16, 16, 2), dim3(256), 0, stream>>>(qb, kb, vTb, ab);

    k_gemm<3><<<gg, dim3(256), 0, stream>>>(ab, wot, bo, out);
}
